// MAE3DWithAttn_3058016715305
// MI455X (gfx1250) — matrix operand order repaired and checked by reading, not compiled, not run on hardware
//
#include <hip/hip_runtime.h>

#define Bc     8
#define Nc     512
#define PDc    512
#define Ec     768
#define Hc     12
#define HDc    64
#define FFc    3072
#define DEPTHc 8

typedef __bf16 bf16_t;
typedef __attribute__((ext_vector_type(16))) __bf16 bf16x16;
typedef __attribute__((ext_vector_type(8)))  float  floatx8;

// ------------------------------------------------------------------
// elementwise helpers
// ------------------------------------------------------------------
__global__ void k_cast_bf16(const float* __restrict__ s, bf16_t* __restrict__ d, long long n) {
  long long i  = (long long)blockIdx.x * blockDim.x + threadIdx.x;
  long long st = (long long)gridDim.x * blockDim.x;
  for (; i < n; i += st) d[i] = (bf16_t)s[i];
}

__global__ void k_add_pos(float* __restrict__ x, const float* __restrict__ pos,
                          bf16_t* __restrict__ xb) {
  long long i = (long long)blockIdx.x * blockDim.x + threadIdx.x;
  if (i >= (long long)Bc * Nc * Ec) return;
  float v = x[i] + pos[i % ((long long)Nc * Ec)];
  x[i]  = v;
  xb[i] = (bf16_t)v;
}

// qkv [B,N,3E] fp32 -> q,k bf16 [B,H,N,HD], vT bf16 [B,H,HD,N]
__global__ void k_repack_qkv(const float* __restrict__ qkv, bf16_t* __restrict__ q,
                             bf16_t* __restrict__ k, bf16_t* __restrict__ vT) {
  long long i = (long long)blockIdx.x * blockDim.x + threadIdx.x;
  if (i >= (long long)Bc * Nc * Ec) return;
  int e       = (int)(i % Ec);
  long long bn = i / Ec;
  int n = (int)(bn % Nc);
  int b = (int)(bn / Nc);
  int h = e / HDc, d = e % HDc;
  const float* src = qkv + bn * (3 * Ec);
  long long bh = (long long)b * Hc + h;
  q [(bh * Nc + n) * HDc + d] = (bf16_t)src[e];
  k [(bh * Nc + n) * HDc + d] = (bf16_t)src[Ec + e];
  vT[(bh * HDc + d) * Nc + n] = (bf16_t)src[2 * Ec + e];
}

// o [B,H,N,HD] fp32 -> merged [B,N,E] bf16
__global__ void k_merge_heads(const float* __restrict__ o, bf16_t* __restrict__ ob) {
  long long i = (long long)blockIdx.x * blockDim.x + threadIdx.x;
  if (i >= (long long)Bc * Hc * Nc * HDc) return;
  int d = (int)(i % HDc);
  long long t = i / HDc;
  int n = (int)(t % Nc); t /= Nc;
  int h = (int)(t % Hc);
  int b = (int)(t / Hc);
  ob[((long long)b * Nc + n) * Ec + h * HDc + d] = (bf16_t)o[i];
}

// in-place softmax over rows of 512; also write bf16 copy
__global__ void k_softmax(float* __restrict__ p, bf16_t* __restrict__ pb, float scale) {
  __shared__ float red[256];
  long long row = blockIdx.x;
  float* r = p + row * (long long)Nc;
  int t = threadIdx.x;
  float a0 = r[t] * scale, a1 = r[t + 256] * scale;
  red[t] = fmaxf(a0, a1);
  __syncthreads();
  for (int s = 128; s > 0; s >>= 1) { if (t < s) red[t] = fmaxf(red[t], red[t + s]); __syncthreads(); }
  float mx = red[0];
  __syncthreads();
  float e0 = __expf(a0 - mx), e1 = __expf(a1 - mx);
  red[t] = e0 + e1;
  __syncthreads();
  for (int s = 128; s > 0; s >>= 1) { if (t < s) red[t] += red[t + s]; __syncthreads(); }
  float inv = 1.0f / red[0];
  e0 *= inv; e1 *= inv;
  r[t] = e0; r[t + 256] = e1;
  bf16_t* rb = pb + row * (long long)Nc;
  rb[t] = (bf16_t)e0; rb[t + 256] = (bf16_t)e1;
}

// y = LN(a + b) * gamma + beta ; writes fp32 (next residual) + bf16 (next GEMM A)
__global__ void k_resid_ln(const float* __restrict__ a, const float* __restrict__ br,
                           const float* __restrict__ gamma, const float* __restrict__ beta,
                           float* __restrict__ of, bf16_t* __restrict__ ob) {
  __shared__ float red[256];
  long long row = blockIdx.x;
  const float* pa = a  + row * (long long)Ec;
  const float* pb = br + row * (long long)Ec;
  int t = threadIdx.x;
  float v[3]; float s = 0.f;
  #pragma unroll
  for (int i = 0; i < 3; ++i) { int e = t + i * 256; v[i] = pa[e] + pb[e]; s += v[i]; }
  red[t] = s; __syncthreads();
  for (int st = 128; st > 0; st >>= 1) { if (t < st) red[t] += red[t + st]; __syncthreads(); }
  float mean = red[0] * (1.0f / Ec);
  __syncthreads();
  float sq = 0.f;
  #pragma unroll
  for (int i = 0; i < 3; ++i) { float d = v[i] - mean; sq += d * d; }
  red[t] = sq; __syncthreads();
  for (int st = 128; st > 0; st >>= 1) { if (t < st) red[t] += red[t + st]; __syncthreads(); }
  float inv = rsqrtf(red[0] * (1.0f / Ec) + 1e-5f);
  #pragma unroll
  for (int i = 0; i < 3; ++i) {
    int e = t + i * 256;
    float y = (v[i] - mean) * inv * gamma[e] + beta[e];
    of[row * (long long)Ec + e] = y;
    ob[row * (long long)Ec + e] = (bf16_t)y;
  }
}

__global__ void k_mean_pool(const float* __restrict__ x, float* __restrict__ out) {
  int i = blockIdx.x * blockDim.x + threadIdx.x;
  if (i >= Bc * Ec) return;
  int b = i / Ec, e = i % Ec;
  const float* p = x + (long long)b * Nc * Ec + e;
  float s = 0.f;
  for (int n = 0; n < Nc; ++n) s += p[(long long)n * Ec];
  out[i] = s * (1.0f / Nc);
}

// ------------------------------------------------------------------
// WMMA GEMM: C[m,n] = sum_k A[m,k] * Bm[n,k]  (C = A * Bm^T), batched via grid.z
// block = 256 threads = 8 waves; block tile 128x128; wave tile 32x64; K-step 32
// double-buffered LDS fed by CDNA5 async global->LDS (ASYNCcnt) loads
// ------------------------------------------------------------------
#define BM 128
#define BN 128
#define BK 32
#define LP 40   // padded LDS row stride (elements): 80B, 16B-aligned, conflict-free

// one 16B async copy: global -> LDS (no VGPR staging; tracked by ASYNCcnt)
__device__ __forceinline__ void async_cp16(const bf16_t* g, bf16_t* l) {
  unsigned loff = (unsigned)(uintptr_t)l;   // low 32 bits of flat ptr = LDS offset
  asm volatile("global_load_async_to_lds_b128 %0, %1, off"
               :: "v"(loff), "v"(g) : "memory");
}

// stage one 128x32 A-tile + B-tile via async loads (4 async ops per wave)
__device__ __forceinline__ void stage_async_tile(
    const bf16_t* __restrict__ A, const bf16_t* __restrict__ Bm,
    bf16_t* As, bf16_t* Bs,
    int m0, int n0, int M, int Nn, int lda, int ldb, int k0, int tid)
{
  #pragma unroll
  for (int s = 0; s < 2; ++s) {
    int idx = (tid + s * 256) * 8;     // element index within 128x32 tile
    int r = idx >> 5;
    int c = idx & 31;
    int ra = m0 + r; if (ra >= M)  ra = M - 1;
    int rb = n0 + r; if (rb >= Nn) rb = Nn - 1;
    async_cp16(A  + (long long)ra * lda + k0 + c, As + r * LP + c);
    async_cp16(Bm + (long long)rb * ldb + k0 + c, Bs + r * LP + c);
  }
}

__global__ __launch_bounds__(256) void k_wmma_gemm(
    const bf16_t* __restrict__ A, const bf16_t* __restrict__ Bm,
    const float* __restrict__ bias,
    float* __restrict__ C, bf16_t* __restrict__ Cbf,
    int M, int Nn, int K, int lda, int ldb, int ldc,
    long long sA, long long sB, long long sC,
    float scale, int relu)
{
  __shared__ bf16_t As[2][BM * LP];
  __shared__ bf16_t Bs[2][BN * LP];

  A  += (long long)blockIdx.z * sA;
  Bm += (long long)blockIdx.z * sB;
  long long coff = (long long)blockIdx.z * sC;

  int m0 = blockIdx.x * BM;
  int n0 = blockIdx.y * BN;

  int tid  = threadIdx.x;
  int wave = tid >> 5;
  int lane = tid & 31;
  int wm   = wave & 3;        // wave row  (0..3) -> 32 rows each
  int wn   = wave >> 2;       // wave col  (0..1) -> 64 cols each
  int hi   = lane >> 4;       // half-wave select
  int lcol = lane & 15;

  floatx8 acc[2][4];
  #pragma unroll
  for (int mi = 0; mi < 2; ++mi)
    #pragma unroll
    for (int ni = 0; ni < 4; ++ni) {
      floatx8 z = {0.f, 0.f, 0.f, 0.f, 0.f, 0.f, 0.f, 0.f};
      acc[mi][ni] = z;
    }

  const int nt = K / BK;
  // prologue: async-stage tile 0 into buffer 0
  stage_async_tile(A, Bm, As[0], Bs[0], m0, n0, M, Nn, lda, ldb, 0, tid);

  for (int kt = 0; kt < nt; ++kt) {
    int cur = kt & 1;
    if (kt + 1 < nt) {
      // issue next tile into the other buffer (its readers finished at the
      // trailing barrier of iteration kt-1), then wait for tile kt only:
      // 4 async ops of tile kt+1 may remain in flight (in-order completion).
      stage_async_tile(A, Bm, As[1 - cur], Bs[1 - cur],
                       m0, n0, M, Nn, lda, ldb, (kt + 1) * BK, tid);
      asm volatile("s_wait_asynccnt 0x4" ::: "memory");
    } else {
      asm volatile("s_wait_asynccnt 0x0" ::: "memory");
    }
    __syncthreads();   // all waves' tile-kt LDS writes visible

    // ---- fragments (ISA 16-bit A 16x32 / B 32x16 lane layouts) ----
    bf16x16 fa[2], fb[4];
    #pragma unroll
    for (int mi = 0; mi < 2; ++mi) {
      const bf16_t* p = &As[cur][(wm * 32 + mi * 16 + lcol) * LP + hi * 8];
      uint4* d = (uint4*)&fa[mi];
      d[0] = *(const uint4*)(p);        // K = hi*8 .. +7
      d[1] = *(const uint4*)(p + 16);   // K = 16+hi*8 .. +7
    }
    #pragma unroll
    for (int ni = 0; ni < 4; ++ni) {
      const bf16_t* p = &Bs[cur][(wn * 64 + ni * 16 + lcol) * LP + hi * 8];
      uint4* d = (uint4*)&fb[ni];
      d[0] = *(const uint4*)(p);        // K = hi*8 .. +7
      d[1] = *(const uint4*)(p + 16);   // K = 16+hi*8 .. +7
    }
    #pragma unroll
    for (int mi = 0; mi < 2; ++mi)
      #pragma unroll
      for (int ni = 0; ni < 4; ++ni)
        acc[mi][ni] = __builtin_amdgcn_wmma_f32_16x16x32_bf16(
            false, fa[mi], false, fb[ni], (short)0, acc[mi][ni], false, false);
    __syncthreads();   // reads of buf[cur] done before it is re-staged
  }

  // ---- epilogue: bias / relu / scale, fp32 and/or bf16 stores ----
  #pragma unroll
  for (int mi = 0; mi < 2; ++mi) {
    #pragma unroll
    for (int ni = 0; ni < 4; ++ni) {
      int n = n0 + wn * 64 + ni * 16 + lcol;
      if (n >= Nn) continue;
      float bv = bias ? bias[n] : 0.f;
      #pragma unroll
      for (int r = 0; r < 8; ++r) {
        int m = m0 + wm * 32 + mi * 16 + hi * 8 + r;
        if (m >= M) continue;
        float v = acc[mi][ni][r] * scale + bv;
        if (relu) v = v > 0.f ? v : 0.f;
        long long o = coff + (long long)m * ldc + n;
        if (C)   C[o]   = v;
        if (Cbf) Cbf[o] = (bf16_t)v;
      }
    }
  }
}

// ------------------------------------------------------------------
// host
// ------------------------------------------------------------------
extern "C" void kernel_launch(void* const* d_in, const int* in_sizes, int n_in,
                              void* d_out, int out_size, void* d_ws, size_t ws_size,
                              hipStream_t stream) {
  (void)in_sizes; (void)n_in; (void)out_size; (void)ws_size;
  const float* patches = (const float*)d_in[0];
  const float* Wp    = (const float*)d_in[1];
  const float* bp    = (const float*)d_in[2];
  const float* pos   = (const float*)d_in[3];
  const float* qkv_w = (const float*)d_in[4];
  const float* qkv_b = (const float*)d_in[5];
  const float* out_w = (const float*)d_in[6];
  const float* out_b = (const float*)d_in[7];
  const float* ln1s  = (const float*)d_in[8];
  const float* ln1b  = (const float*)d_in[9];
  const float* w1    = (const float*)d_in[10];
  const float* b1    = (const float*)d_in[11];
  const float* w2    = (const float*)d_in[12];
  const float* b2    = (const float*)d_in[13];
  const float* ln2s  = (const float*)d_in[14];
  const float* ln2b  = (const float*)d_in[15];

  char* base = (char*)d_ws; size_t off = 0;
  auto alloc = [&](size_t bytes) -> void* {
    void* p = base + off; off = (off + bytes + 255) & ~(size_t)255; return p;
  };

  bf16_t* patches_bf = (bf16_t*)alloc((size_t)Bc*Nc*PDc*2);
  bf16_t* Wp_bf   = (bf16_t*)alloc((size_t)Ec*PDc*2);
  bf16_t* qkvw_bf = (bf16_t*)alloc((size_t)DEPTHc*3*Ec*Ec*2);
  bf16_t* outw_bf = (bf16_t*)alloc((size_t)DEPTHc*Ec*Ec*2);
  bf16_t* w1_bf   = (bf16_t*)alloc((size_t)DEPTHc*FFc*Ec*2);
  bf16_t* w2_bf   = (bf16_t*)alloc((size_t)DEPTHc*Ec*FFc*2);
  float*  x_f     = (float*) alloc((size_t)Bc*Nc*Ec*4);
  bf16_t* x_bf    = (bf16_t*)alloc((size_t)Bc*Nc*Ec*2);
  float*  x2_f    = (float*) alloc((size_t)Bc*Nc*Ec*4);
  bf16_t* x2_bf   = (bf16_t*)alloc((size_t)Bc*Nc*Ec*2);
  float*  qkv_f   = (float*) alloc((size_t)Bc*Nc*3*Ec*4);
  bf16_t* q_bf    = (bf16_t*)alloc((size_t)Bc*Hc*Nc*HDc*2);
  bf16_t* k_bf    = (bf16_t*)alloc((size_t)Bc*Hc*Nc*HDc*2);
  bf16_t* vT_bf   = (bf16_t*)alloc((size_t)Bc*Hc*HDc*Nc*2);
  bf16_t* attn_bf = (bf16_t*)alloc((size_t)Bc*Hc*Nc*Nc*2);
  float*  o_f     = (float*) alloc((size_t)Bc*Hc*Nc*HDc*4);
  bf16_t* o_bf    = (bf16_t*)alloc((size_t)Bc*Nc*Ec*2);
  float*  proj_f  = (float*) alloc((size_t)Bc*Nc*Ec*4);
  bf16_t* ff1_bf  = (bf16_t*)alloc((size_t)Bc*Nc*FFc*2);
  float*  ff2_f   = (float*) alloc((size_t)Bc*Nc*Ec*4);

  auto cast = [&](const float* s, bf16_t* d, long long n) {
    int blocks = (int)((n + 255) / 256); if (blocks > 2048) blocks = 2048;
    k_cast_bf16<<<blocks, 256, 0, stream>>>(s, d, n);
  };
  auto gemm = [&](const bf16_t* A, const bf16_t* Bmat, const float* bias,
                  float* C, bf16_t* Cbf, int M, int Nn, int K,
                  int lda, int ldb, int ldc,
                  long long sA, long long sB, long long sC, int batch,
                  float scale, int relu) {
    dim3 g((M + BM - 1) / BM, (Nn + BN - 1) / BN, batch);
    k_wmma_gemm<<<g, dim3(256), 0, stream>>>(A, Bmat, bias, C, Cbf,
                                             M, Nn, K, lda, ldb, ldc,
                                             sA, sB, sC, scale, relu);
  };

  // ---- one-time (per launch) bf16 weight casts ----
  cast(patches, patches_bf, (long long)Bc*Nc*PDc);
  cast(Wp,    Wp_bf,   (long long)Ec*PDc);
  cast(qkv_w, qkvw_bf, (long long)DEPTHc*3*Ec*Ec);
  cast(out_w, outw_bf, (long long)DEPTHc*Ec*Ec);
  cast(w1,    w1_bf,   (long long)DEPTHc*FFc*Ec);
  cast(w2,    w2_bf,   (long long)DEPTHc*Ec*FFc);

  const int M  = Bc * Nc;                 // 4096
  const int BH = Bc * Hc;                 // 96
  const long long NE = (long long)Bc*Nc*Ec;

  // ---- patch embed + pos ----
  gemm(patches_bf, Wp_bf, bp, x_f, nullptr, M, Ec, PDc, PDc, PDc, Ec, 0, 0, 0, 1, 1.f, 0);
  k_add_pos<<<(int)((NE + 255) / 256), 256, 0, stream>>>(x_f, pos, x_bf);

  float* dout = (float*)d_out;
  for (int l = 0; l < DEPTHc; ++l) {
    // qkv projection
    gemm(x_bf, qkvw_bf + (size_t)l*3*Ec*Ec, qkv_b + (size_t)l*3*Ec,
         qkv_f, nullptr, M, 3*Ec, Ec, Ec, Ec, 3*Ec, 0, 0, 0, 1, 1.f, 0);
    k_repack_qkv<<<(int)((NE + 255) / 256), 256, 0, stream>>>(qkv_f, q_bf, k_bf, vT_bf);

    // scores -> written directly into d_out attn slice (fp32), softmax in place
    float* attn = dout + 6144 + (long long)l * Bc * Hc * Nc * Nc;
    gemm(q_bf, k_bf, nullptr, attn, nullptr, Nc, Nc, HDc, HDc, HDc, Nc,
         (long long)Nc*HDc, (long long)Nc*HDc, (long long)Nc*Nc, BH, 1.f, 0);
    k_softmax<<<BH * Nc, 256, 0, stream>>>(attn, attn_bf, 0.125f);

    // attn @ v  (B operand = vT rows contiguous)
    gemm(attn_bf, vT_bf, nullptr, o_f, nullptr, Nc, HDc, Nc, Nc, Nc, HDc,
         (long long)Nc*Nc, (long long)HDc*Nc, (long long)Nc*HDc, BH, 1.f, 0);
    k_merge_heads<<<(int)((NE + 255) / 256), 256, 0, stream>>>(o_f, o_bf);

    // output projection
    gemm(o_bf, outw_bf + (size_t)l*Ec*Ec, out_b + (size_t)l*Ec,
         proj_f, nullptr, M, Ec, Ec, Ec, Ec, Ec, 0, 0, 0, 1, 1.f, 0);

    // x2 = LN1(x + proj)
    k_resid_ln<<<M, 256, 0, stream>>>(x_f, proj_f, ln1s + (size_t)l*Ec, ln1b + (size_t)l*Ec,
                                      x2_f, x2_bf);

    // FFN: relu(x2 @ W1^T + b1) @ W2^T + b2
    gemm(x2_bf, w1_bf + (size_t)l*FFc*Ec, b1 + (size_t)l*FFc,
         nullptr, ff1_bf, M, FFc, Ec, Ec, Ec, FFc, 0, 0, 0, 1, 1.f, 1);
    gemm(ff1_bf, w2_bf + (size_t)l*Ec*FFc, b2 + (size_t)l*Ec,
         ff2_f, nullptr, M, Ec, FFc, FFc, FFc, Ec, 0, 0, 0, 1, 1.f, 0);

    // x = LN2(x2 + ff2)
    k_resid_ln<<<M, 256, 0, stream>>>(x2_f, ff2_f, ln2s + (size_t)l*Ec, ln2b + (size_t)l*Ec,
                                      x_f, x_bf);
  }

  // pooled mean over sequence -> d_out[0 : B*E]
  k_mean_pool<<<(Bc * Ec + 255) / 256, 256, 0, stream>>>(x_f, dout);
}
